// WordSequence_2628519985197
// MI455X (gfx1250) — matrix operand order repaired and checked by reading, not compiled, not run on hardware
//
#include <hip/hip_runtime.h>
#include <hip/hip_bf16.h>
#include <math.h>

// Problem dims (fixed by reference)
#define Bb 32
#define Ss 256
#define Dd 128
#define Gg 128
#define Hh 256      // total hidden
#define HH 128      // per-direction hidden
#define Ii 256      // LSTM input = D + G
#define Ll 32
#define BS (Bb*Ss)  // 8192
#define NG 512      // 4 * HH gate width

typedef __attribute__((ext_vector_type(16))) _Float16 v16h;
typedef __attribute__((ext_vector_type(8)))  float    v8f;
typedef __attribute__((ext_vector_type(4)))  unsigned int u32x4;
typedef __attribute__((ext_vector_type(8)))  int      i32x8;
typedef __attribute__((ext_vector_type(4)))  int      i32x4;

// ---------------- WMMA fragment helpers (CDNA5 16x16x32 f16) ----------------
// A (16x32, row-major source, stride lda in halves):
//   lane<16 : h[0..7]=K0..7,  h[8..15]=K16..23   (M = lane&15)
//   lane>=16: h[0..7]=K8..15, h[8..15]=K24..31
// -> two contiguous 16B segments per lane: [k0+8*grp .. +7] and [k0+8*grp+16 .. +7]
__device__ __forceinline__ v16h load_frag_a(const _Float16* A, int lda,
                                            int m0, int k0, int lane) {
  int m   = m0 + (lane & 15);
  int grp = (lane >> 4) & 1;
  const _Float16* p = A + (size_t)m * lda + k0 + 8 * grp;
  v16h a;
  ((uint4*)&a)[0] = *(const uint4*)(p);
  ((uint4*)&a)[1] = *(const uint4*)(p + 16);
  return a;
}

// Packed (fragment-major) B: frag f, lane l -> 16 halves contiguous at (f*32+l)*16
__device__ __forceinline__ v16h load_frag_packed(const _Float16* P, int frag, int lane) {
  const uint4* p = (const uint4*)(P + (((size_t)frag * 32 + lane) << 4));
  v16h r;
  ((uint4*)&r)[0] = p[0];
  ((uint4*)&r)[1] = p[1];
  return r;
}

__device__ __forceinline__ v8f wmma_f16(v16h a, v16h b, v8f c) {
  return __builtin_amdgcn_wmma_f32_16x16x32_f16(false, a, false, b,
                                                (short)0, c, false, false);
}

__device__ __forceinline__ float sigf(float x) { return 1.f / (1.f + __expf(-x)); }

// ---------------- pack f32 weight (K x N row-major) into WMMA B fragments ----
// frag layout: B[16x(lane&15) column, K interleaved 8/8 across lane halves]:
//   lanes 0-15 hold K=k0..k0+7, k0+16..k0+23; lanes 16-31 hold K=k0+8..k0+15, k0+24..k0+31
__global__ void k_packw(const float* __restrict__ src, _Float16* __restrict__ dst,
                        int K, int N) {
  int tile = blockIdx.x, lane = threadIdx.x;
  int NT = N >> 4;
  int kt = tile / NT, nt = tile % NT;
  int n     = nt * 16 + (lane & 15);
  int kbase = kt * 32 + ((lane >> 4) & 1) * 8;
  _Float16* d = dst + (((size_t)tile * 32 + lane) << 4);
#pragma unroll
  for (int j = 0; j < 16; ++j) d[j] = (_Float16)src[(size_t)(kbase + j + (j >= 8 ? 8 : 0)) * N + n];
}

// ---------------- label embedding row-normalize (32 x 128) ----------------
__global__ void k_lnorm(const float* __restrict__ lemb, float* __restrict__ ln) {
  __shared__ float red[Dd];
  int l = blockIdx.x, d = threadIdx.x;
  float v = lemb[l * Dd + d];
  red[d] = v * v; __syncthreads();
  for (int off = 64; off > 0; off >>= 1) {
    if (d < off) red[d] += red[d + off];
    __syncthreads();
  }
  ln[l * Dd + d] = v / (sqrtf(red[0]) + 1e-8f);
}

// ---------------- embed gather + sent_hidden + label_embs ----------------
__global__ void k_embed(const int* __restrict__ word, const float* __restrict__ wemb,
                        const float* __restrict__ sentW, const float* __restrict__ sentb,
                        const float* __restrict__ ln,
                        float* __restrict__ we, float* __restrict__ sh,
                        float* __restrict__ lab) {
  __shared__ float wel[Dd], wn[Dd], red[Dd];
  int bs = blockIdx.x, d = threadIdx.x;
  int wi = word[bs];
  float v = wemb[(size_t)wi * Dd + d];
  wel[d] = v;
  we[(size_t)bs * Dd + d] = v;
  red[d] = v * v; __syncthreads();
  for (int off = 64; off > 0; off >>= 1) {
    if (d < off) red[d] += red[d + off];
    __syncthreads();
  }
  float inv = 1.f / (sqrtf(red[0]) + 1e-8f);
  wn[d] = v * inv; __syncthreads();
  float acc = sentb[d];
  for (int k = 0; k < Dd; ++k) acc += wel[k] * sentW[k * Gg + d];
  sh[(size_t)bs * Gg + d] = tanhf(acc);
  if (d < Ll) {
    const float* lr = ln + d * Dd;
    float a = 0.f;
    for (int k = 0; k < Dd; ++k) a += wn[k] * lr[k];
    lab[(size_t)bs * Ll + d] = a;
  }
}

// ---------------- 1D conv (K=3, same pad) + relu + channel-max ----------------
__global__ void k_conv(const float* __restrict__ lab, const float* __restrict__ convW,
                       const float* __restrict__ convb, float* __restrict__ lm) {
  __shared__ float r32[Ll];
  int bs = blockIdx.x, b = bs >> 8, s = bs & 255;
  int o = threadIdx.x;
  float acc = convb[o];
#pragma unroll
  for (int k = 0; k < 3; ++k) {
    int s2 = s + k - 1;
    if (s2 >= 0 && s2 < Ss) {
      const float* lr = lab + (size_t)(b * Ss + s2) * Ll;
      const float* wr = convW + (size_t)o * Ll * 3 + k;
      for (int i = 0; i < Ll; ++i) acc += lr[i] * wr[i * 3];
    }
  }
  acc = fmaxf(acc, 0.f);
  r32[o] = acc; __syncthreads();
  for (int off = 16; off > 0; off >>= 1) {
    if (o < off) r32[o] = fmaxf(r32[o], r32[o + off]);
    __syncthreads();
  }
  if (o == 0) lm[bs] = r32[0];
}

// ---------------- masked softmax attention + sent_rep ----------------
__global__ void k_attn(const float* __restrict__ lm, const int* __restrict__ lens,
                       const float* __restrict__ sh, float* __restrict__ srep) {
  __shared__ float at[Ss], red[Ss];
  int b = blockIdx.x, s = threadIdx.x;
  float e = (s < lens[b]) ? __expf(lm[b * Ss + s]) : 0.f;
  red[s] = e; __syncthreads();
  for (int off = 128; off > 0; off >>= 1) {
    if (s < off) red[s] += red[s + off];
    __syncthreads();
  }
  at[s] = e / red[0]; __syncthreads();
  if (s < Gg) {
    float acc = 0.f;
    for (int t = 0; t < Ss; ++t) acc += at[t] * sh[((size_t)b * Ss + t) * Gg + s];
    srep[b * Gg + s] = acc;
  }
}

// ---------------- build f16 wr = [we | sent_rep] ----------------
__global__ void k_buildwr(const float* __restrict__ we, const float* __restrict__ srep,
                          _Float16* __restrict__ wrh) {
  int bs = blockIdx.x, i = threadIdx.x, b = bs >> 8;
  float v = (i < Dd) ? we[(size_t)bs * Dd + i] : srep[b * Gg + (i - Dd)];
  wrh[(size_t)bs * Ii + i] = (_Float16)v;
}

// ---------------- big WMMA GEMM: xW = wr(8192x256) @ W(256x512) + bias ----------------
// block = 256 threads (8 waves); per block: 32x64 output tile (2 M-tiles x 4 N-tiles)
__global__ void k_gemm_xw(const _Float16* __restrict__ wrh,
                          const _Float16* __restrict__ Wfp, const _Float16* __restrict__ Wbp,
                          const float* __restrict__ bf, const float* __restrict__ bb,
                          float* __restrict__ xWf, float* __restrict__ xWb) {
  const int lane = threadIdx.x & 31, w = threadIdx.x >> 5;
  const _Float16* Wm; const float* bias; float* out;
  if (blockIdx.z == 0) { Wm = Wfp; bias = bf; out = xWf; }
  else                 { Wm = Wbp; bias = bb; out = xWb; }
  int m0  = blockIdx.x * 32 + (w & 1) * 16;
  int n0  = blockIdx.y * 64 + (w >> 1) * 16;
  int ntg = n0 >> 4;                 // global n-tile for packed weights (NT=32)
  v8f acc = {};
#pragma unroll
  for (int k0 = 0; k0 < Ii; k0 += 32) {
    v16h a = load_frag_a(wrh, Ii, m0, k0, lane);
    v16h b = load_frag_packed(Wm, (k0 >> 5) * 32 + ntg, lane);
    acc = wmma_f16(a, b, acc);
  }
  int n = n0 + (lane & 15);
  float bn = bias[n];
#pragma unroll
  for (int r = 0; r < 8; ++r) {
    int m = m0 + r + 8 * (lane >> 4);
    out[(size_t)m * NG + n] = acc[r] + bn;
  }
}

// ---------------- persistent LSTM recurrence: one workgroup per direction ----------------
// gates(32x512) = h(32x128, f16 LDS) @ U(128x512, f16 packed in LDS) + xW(t)
// wave w: rows 16*(w&1)..+15, columns (w>>1)*32..+31 of ALL 4 gates -> each lane owns its
// i/f/g/o quadruple and 16 register-resident cells c.
__global__ void k_lstm(const float* __restrict__ xWf, const float* __restrict__ xWb,
                       const _Float16* __restrict__ Ufp, const _Float16* __restrict__ Ubp,
                       const int* __restrict__ lens,
                       float* __restrict__ hf, float* __restrict__ hb) {
  __shared__ _Float16 h_sh[Bb * HH];        // 8 KB hidden state, f16
  extern __shared__ _Float16 Up[];          // 128 KB packed recurrent weights
  const int dir = blockIdx.x;
  const float*    xW   = dir ? xWb : xWf;
  const _Float16* Ug   = dir ? Ubp : Ufp;
  float*          hout = dir ? hb  : hf;
  const int tid = threadIdx.x, lane = tid & 31, w = tid >> 5;
  const int m0 = (w & 1) * 16;
  const int npair = w >> 1;                 // 0..3 -> columns npair*32..+31
  for (int i = tid; i < Bb * HH; i += 256) h_sh[i] = (_Float16)0.f;

  // ---- stage packed U (128 KB) into LDS: TDM if available, else b128 copy ----
#if __has_builtin(__builtin_amdgcn_tensor_load_to_lds) && __has_builtin(__builtin_amdgcn_s_wait_tensorcnt)
  if (tid < 32) {  // one wave issues the DMA (TDM ignores EXEC; per-wave op)
    unsigned ldsA = (unsigned)(uintptr_t)(void*)Up;               // LDS offset
    unsigned long long ga = (unsigned long long)(uintptr_t)Ug;
    // D# group0: count=1, lds_addr, 57b global_addr, type=2 ("image")
    u32x4 g0 = { 1u, ldsA,
                 (unsigned)(ga & 0xffffffffu),
                 ((unsigned)((ga >> 32) & 0x01ffffffu)) | (2u << 30) };
    // D# group1: data_size=2B; 2D tensor 4096x16, tile 4096x16, stride0=4096
    i32x8 g1 = { (int)0x00010000,        // wg_mask=0, data_size=1 (2 bytes)
                 (int)(4096u << 16),     // tensor_dim0[15:0]=4096 (bits 63:48)
                 (int)(16u << 16),       // tensor_dim0[31:16]=0 | tensor_dim1[15:0]=16
                 (int)(4096u << 16),     // tensor_dim1[31:16]=0 | tile_dim0=4096
                 16,                     // tile_dim1=16, tile_dim2=0
                 4096,                   // tensor_dim0_stride[31:0]
                 0, 0 };
    i32x4 z4 = { 0, 0, 0, 0 };
#if __clang_major__ >= 23
    i32x8 z8 = { 0, 0, 0, 0, 0, 0, 0, 0 };
    __builtin_amdgcn_tensor_load_to_lds(g0, g1, z4, z4, z8, 0);
#else
    __builtin_amdgcn_tensor_load_to_lds(g0, g1, z4, z4, 0);
#endif
    __builtin_amdgcn_s_wait_tensorcnt(0);
  }
#else
  {
    const uint4* s = (const uint4*)Ug;
    uint4*       d = (uint4*)Up;
    for (int i = tid; i < (HH * NG) / 8; i += 256) d[i] = s[i];
  }
#endif
  float c[2][8] = {};
  __syncthreads();

  for (int step = 0; step < Ss; ++step) {
    int t = dir ? (Ss - 1 - step) : step;
    // prefetch next step's gate pre-activations while we do the matmul
    int tn = dir ? (t - 1) : (t + 1);
    if (tn >= 0 && tn < Ss) {
      int mpf = m0 + (lane & 15);
      __builtin_prefetch(xW + ((size_t)mpf * Ss + tn) * NG + npair * 32, 0, 1);
    }
    v8f acc[4][2] = {};
#pragma unroll
    for (int k0 = 0; k0 < HH; k0 += 32) {
      v16h a = load_frag_a(h_sh, HH, m0, k0, lane);            // ds_load_b128 x2
#pragma unroll
      for (int g = 0; g < 4; ++g)
#pragma unroll
        for (int nt = 0; nt < 2; ++nt) {
          int ntg = (g * HH + npair * 32 + nt * 16) >> 4;
          v16h b = load_frag_packed(Up, (k0 >> 5) * 32 + ntg, lane); // ds_load_b128 x2
          acc[g][nt] = wmma_f16(a, b, acc[g][nt]);
        }
    }
    __syncthreads();   // all waves done reading h_sh for this step
#pragma unroll
    for (int nt = 0; nt < 2; ++nt) {
      int col = npair * 32 + nt * 16 + (lane & 15);
#pragma unroll
      for (int r = 0; r < 8; ++r) {
        int m = m0 + r + 8 * (lane >> 4);                      // batch row 0..31
        const float* xwrow = xW + ((size_t)m * Ss + t) * NG;
        float gi = acc[0][nt][r] + xwrow[col];
        float gf = acc[1][nt][r] + xwrow[HH + col];
        float gg = acc[2][nt][r] + xwrow[2 * HH + col];
        float go = acc[3][nt][r] + xwrow[3 * HH + col];
        float si = sigf(gi), sf = sigf(gf), so = sigf(go);
        float tg = tanhf(gg);
        float cn = sf * c[nt][r] + si * tg;
        float hn = so * tanhf(cn);
        bool msk = (t < lens[m]);
        if (msk) { c[nt][r] = cn; h_sh[m * HH + col] = (_Float16)hn; }
        hout[((size_t)m * Ss + t) * HH + col] = msk ? hn : 0.f;
      }
    }
    __syncthreads();   // h_sh updated before next step's reads
  }
}

// ---------------- output: feat@tagW + tagb blended with masked mem read ----------------
__global__ void k_out(const float* __restrict__ hf, const float* __restrict__ hb,
                      const float* __restrict__ tagW, const float* __restrict__ tagb,
                      const int* __restrict__ word, const int* __restrict__ lens,
                      const float* __restrict__ membank, const float* __restrict__ memW,
                      const float* __restrict__ memb, float* __restrict__ out) {
  int bs = blockIdx.x, l = threadIdx.x, b = bs >> 8, s = bs & 255;
  float a1 = tagb[l];
  const float* hfr = hf + (size_t)bs * HH;
  const float* hbr = hb + (size_t)bs * HH;
  for (int d = 0; d < HH; ++d) a1 += hfr[d] * tagW[d * Ll + l];
  for (int d = 0; d < HH; ++d) a1 += hbr[d] * tagW[(HH + d) * Ll + l];
  float msk = (s < lens[b]) ? 1.f : 0.f;
  const float* mr = membank + (size_t)word[bs] * Hh;
  float dot = 0.f;
  for (int d = 0; d < Hh; ++d) dot += mr[d] * memW[d * Ll + l];
  float a2 = memb[l] + msk * dot;
  out[(size_t)bs * Ll + l] = 0.5f * a1 + 0.5f * a2;   // ALPHA = 0.5
}

extern "C" void kernel_launch(void* const* d_in, const int* in_sizes, int n_in,
                              void* d_out, int out_size, void* d_ws, size_t ws_size,
                              hipStream_t stream) {
  const int*   word  = (const int*)  d_in[0];
  const int*   lens  = (const int*)  d_in[1];
  // d_in[2..6] unused: mask recomputed from lens; idx only feeds the deleted new_bank.
  const float* wemb  = (const float*)d_in[7];
  const float* lemb  = (const float*)d_in[8];
  const float* sentW = (const float*)d_in[9];
  const float* sentb = (const float*)d_in[10];
  const float* convW = (const float*)d_in[11];
  const float* convb = (const float*)d_in[12];
  const float* Wf    = (const float*)d_in[13];
  const float* Uf    = (const float*)d_in[14];
  const float* bf    = (const float*)d_in[15];
  const float* Wb    = (const float*)d_in[16];
  const float* Ub    = (const float*)d_in[17];
  const float* bb    = (const float*)d_in[18];
  const float* tagW  = (const float*)d_in[19];
  const float* tagb  = (const float*)d_in[20];
  const float* memW  = (const float*)d_in[21];
  const float* memb  = (const float*)d_in[22];
  const float* mbank = (const float*)d_in[23];
  float* out = (float*)d_out;

  // workspace carve (floats, then halves; all 32B-aligned by construction)
  float* ws   = (float*)d_ws;
  float* we   = ws;                      // 8192*128
  float* sh   = we   + BS * Dd;          // 8192*128
  float* ln   = sh   + BS * Gg;          // 32*128
  float* lab  = ln   + Ll * Dd;          // 8192*32
  float* lm   = lab  + BS * Ll;          // 8192
  float* srep = lm   + BS;               // 32*128
  float* xWf  = srep + Bb * Gg;          // 8192*512
  float* xWb  = xWf  + (size_t)BS * NG;
  float* hf   = xWb  + (size_t)BS * NG;  // 8192*128
  float* hb   = hf   + (size_t)BS * HH;
  _Float16* wrh = (_Float16*)(hb + (size_t)BS * HH); // 8192*256
  _Float16* Wfp = wrh + (size_t)BS * Ii;             // 256*512 packed
  _Float16* Wbp = Wfp + Ii * NG;
  _Float16* Ufp = Wbp + Ii * NG;                     // 128*512 packed
  _Float16* Ubp = Ufp + HH * NG;

  k_lnorm  <<<Ll, Dd, 0, stream>>>(lemb, ln);
  k_embed  <<<BS, Dd, 0, stream>>>(word, wemb, sentW, sentb, ln, we, sh, lab);
  k_conv   <<<BS, Ll, 0, stream>>>(lab, convW, convb, lm);
  k_attn   <<<Bb, Ss, 0, stream>>>(lm, lens, sh, srep);
  k_buildwr<<<BS, Ii, 0, stream>>>(we, srep, wrh);
  k_packw  <<<(Ii / 32) * (NG / 16), 32, 0, stream>>>(Wf, Wfp, Ii, NG);
  k_packw  <<<(Ii / 32) * (NG / 16), 32, 0, stream>>>(Wb, Wbp, Ii, NG);
  k_packw  <<<(HH / 32) * (NG / 16), 32, 0, stream>>>(Uf, Ufp, HH, NG);
  k_packw  <<<(HH / 32) * (NG / 16), 32, 0, stream>>>(Ub, Ubp, HH, NG);
  k_gemm_xw<<<dim3(BS / 32, NG / 64, 2), 256, 0, stream>>>(wrh, Wfp, Wbp, bf, bb, xWf, xWb);
  k_lstm   <<<2, 256, (size_t)HH * NG * sizeof(_Float16), stream>>>(xWf, xWb, Ufp, Ubp,
                                                                    lens, hf, hb);
  k_out    <<<BS, Ll, 0, stream>>>(hf, hb, tagW, tagb, word, lens, mbank, memW, memb, out);
}
